// ToyGatedMoE_78632261255538
// MI455X (gfx1250) — compile-verified, matrix operand order checked by reading, not run on hardware
//
#include <hip/hip_runtime.h>
#include <cstdint>

#define NTOK 16384
#define HDIM 1024
#define NEXP 8
#define ROWS 32                 // gathered tokens per workgroup (2 M-tiles)
#define TPE  (NTOK / ROWS)      // 512 max tiles per expert
#define XS   (HDIM + 8)         // padded LDS row stride (ushorts) to spread banks
#define KB   2                  // K blocks
#define KSB  16                 // K-steps per block (KB*KSB*32 == HDIM)
#define NTW  8                  // N-tiles per wave

typedef __attribute__((ext_vector_type(16))) __bf16        v16bf;
typedef __attribute__((ext_vector_type(8)))  float         v8f;
typedef __attribute__((ext_vector_type(4)))  unsigned int  u32x4;

union ABVec { u32x4 q[2]; v16bf v; };

static __device__ __forceinline__ unsigned short f2bf(float f) {
  unsigned int u = __float_as_uint(f);
  u += 0x7fffu + ((u >> 16) & 1u);          // round-to-nearest-even
  return (unsigned short)(u >> 16);
}

// Load a 16x32 bf16 WMMA A/B fragment slice for this lane.
// Lane layout (ISA 7.12.2, 16-bit A 16x32): lanes 0-15 hold K={0..7,16..23},
// lanes 16-31 hold K={8..15,24..31}; caller pre-applies the +8 K offset.
static __device__ __forceinline__ v16bf load_frag(const unsigned short* p) {
  ABVec t;
  t.q[0] = *(const u32x4*)(p);        // K 0..7   (8 bf16)
  t.q[1] = *(const u32x4*)(p + 16);   // K 16..23 (8 bf16)
  return t.v;
}

// ---------------------------------------------------------------- kernels ---

__global__ __launch_bounds__(32) void zero_counts_k(int* __restrict__ counts) {
  if (threadIdx.x < NEXP) counts[threadIdx.x] = 0;
}

// One wave per token: 8 dot products of length 1024, shuffle-reduce, argmax,
// atomic append into the winning expert's token list.
__global__ __launch_bounds__(256) void gate_route_k(
    const float* __restrict__ tokens, const float* __restrict__ gate_w,
    int* __restrict__ counts, int* __restrict__ lists) {
  int wave = (int)((blockIdx.x * blockDim.x + threadIdx.x) >> 5);
  int lane = threadIdx.x & 31;
  if (wave >= NTOK) return;
  const float* tk = tokens + (size_t)wave * HDIM;
  float acc[NEXP];
#pragma unroll
  for (int e = 0; e < NEXP; ++e) acc[e] = 0.f;
  for (int j = lane; j < HDIM; j += 32) {
    float x = __builtin_nontemporal_load(&tk[j]);
#pragma unroll
    for (int e = 0; e < NEXP; ++e) acc[e] += x * gate_w[e * HDIM + j];
  }
#pragma unroll
  for (int e = 0; e < NEXP; ++e) {
#pragma unroll
    for (int off = 16; off > 0; off >>= 1)
      acc[e] += __shfl_xor(acc[e], off, 32);
  }
  if (lane == 0) {
    float best = acc[0]; int be = 0;
#pragma unroll
    for (int e = 1; e < NEXP; ++e)
      if (acc[e] > best) { best = acc[e]; be = e; }   // first max, like argmax
    int slot = atomicAdd(&counts[be], 1);
    lists[be * NTOK + slot] = wave;
  }
}

// f32 -> bf16 weight conversion (both expert weight tensors), 4 elems/thread.
__global__ __launch_bounds__(256) void cvt_weights_k(
    const float* __restrict__ w1, const float* __restrict__ w2,
    unsigned short* __restrict__ w1b, unsigned short* __restrict__ w2b) {
  size_t i = ((size_t)blockIdx.x * blockDim.x + threadIdx.x) * 4;
#pragma unroll
  for (int j = 0; j < 4; ++j) {
    w1b[i + j] = f2bf(__builtin_nontemporal_load(&w1[i + j]));
    w2b[i + j] = f2bf(__builtin_nontemporal_load(&w2[i + j]));
  }
}

// GEMM core for one layer: D(acc) = Xlds @ W^T for this wave's 8 N-tiles and
// both 16-row M-tiles. K is processed in 2 blocks of 16 WMMA steps; each
// block preloads 256 VGPRs of A (both M-tiles), so peak pressure stays
// ~384 VGPRs (no spills) while each B fragment from L2 feeds two WMMAs.
static __device__ __forceinline__ void gemm_layer(
    const unsigned short* __restrict__ sA,   // LDS, ROWS x XS bf16
    const unsigned short* __restrict__ we,   // expert weight, HDIM x HDIM bf16
    int waveid, int col, int koff,
    v8f acc0[NTW], v8f acc1[NTW]) {
#pragma unroll
  for (int i = 0; i < NTW; ++i) { acc0[i] = (v8f){}; acc1[i] = (v8f){}; }
  const unsigned short* aptr0 = sA + (0 * 16 + col) * XS + koff;
  const unsigned short* aptr1 = sA + (1 * 16 + col) * XS + koff;
#pragma unroll
  for (int kb = 0; kb < KB; ++kb) {
    const int kbase = kb * (KSB * 32);
    v16bf a0[KSB], a1[KSB];
#pragma unroll
    for (int k = 0; k < KSB; ++k) {
      a0[k] = load_frag(aptr0 + kbase + 32 * k);
      a1[k] = load_frag(aptr1 + kbase + 32 * k);
    }
#pragma unroll
    for (int i = 0; i < NTW; ++i) {
      const int nt = waveid + 8 * i;          // waves cover adjacent N-tiles
      const unsigned short* bptr =
          we + (size_t)(nt * 16 + col) * HDIM + koff + kbase;
#pragma unroll
      for (int k = 0; k < KSB; ++k) {
        v16bf b = load_frag(bptr + 32 * k);
        acc0[i] = __builtin_amdgcn_wmma_f32_16x16x32_bf16(
            false, a0[k], false, b, (short)0, acc0[i], false, false);
        acc1[i] = __builtin_amdgcn_wmma_f32_16x16x32_bf16(
            false, a1[k], false, b, (short)0, acc1[i], false, false);
      }
    }
  }
}

// One workgroup = one expert * one tile of 32 gathered tokens.
__global__ __launch_bounds__(256) void moe_ffn_k(
    const float* __restrict__ tokens,
    const unsigned short* __restrict__ w1b, const float* __restrict__ b1,
    const unsigned short* __restrict__ w2b, const float* __restrict__ b2,
    const int* __restrict__ lists, const int* __restrict__ counts,
    float* __restrict__ out) {
  __shared__ unsigned short sX[ROWS * XS];   // 64.5 KB
  __shared__ unsigned short sH[ROWS * XS];   // 64.5 KB

  const int e    = blockIdx.x >> 9;          // / TPE
  const int t    = blockIdx.x & (TPE - 1);
  const int cnt  = counts[e];
  const int row0 = t * ROWS;
  if (row0 >= cnt) return;                   // uniform early exit
  const int* lst = lists + e * NTOK;

  // ---- stage gathered X tile (f32 -> bf16) into LDS -----------------------
  for (int i = threadIdx.x; i < ROWS * HDIM; i += 256) {
    int r = i >> 10, c = i & (HDIM - 1);
    int slot = row0 + r;
    float v = 0.f;
    if (slot < cnt)
      v = __builtin_nontemporal_load(&tokens[(size_t)lst[slot] * HDIM + c]);
    sX[r * XS + c] = f2bf(v);
  }
  __syncthreads();

  const int waveid = threadIdx.x >> 5;
  const int lane   = threadIdx.x & 31;
  const int col    = lane & 15;                    // tile row (A) / col (B,C)
  const int koff   = (lane & 16) ? 8 : 0;          // K offset per lane group
  const int mhalf  = (lane & 16) ? 8 : 0;          // C/D row offset
  const unsigned short* w1e = w1b + (size_t)e * HDIM * HDIM;
  const unsigned short* w2e = w2b + (size_t)e * HDIM * HDIM;

  v8f acc0[NTW], acc1[NTW];

  // ---- layer 1: sH = relu(sX @ W1^T + b1) ---------------------------------
  gemm_layer(sX, w1e, waveid, col, koff, acc0, acc1);
#pragma unroll
  for (int i = 0; i < NTW; ++i) {
    const int f0 = (waveid + 8 * i) * 16;
    const float bias = b1[e * HDIM + f0 + col];
#pragma unroll
    for (int j = 0; j < 8; ++j) {
      float v0 = acc0[i][j] + bias;
      float v1 = acc1[i][j] + bias;
      v0 = v0 > 0.f ? v0 : 0.f;
      v1 = v1 > 0.f ? v1 : 0.f;
      sH[(mhalf + j) * XS + f0 + col]      = f2bf(v0);
      sH[(16 + mhalf + j) * XS + f0 + col] = f2bf(v1);
    }
  }
  __syncthreads();

  // ---- layer 2: out[tok] = sH @ W2^T + b2 (scatter) -----------------------
  gemm_layer(sH, w2e, waveid, col, koff, acc0, acc1);
#pragma unroll
  for (int i = 0; i < NTW; ++i) {
    const int f0 = (waveid + 8 * i) * 16;
    const float bias = b2[e * HDIM + f0 + col];
#pragma unroll
    for (int j = 0; j < 8; ++j) {
      int slot0 = row0 + mhalf + j;
      int slot1 = row0 + 16 + mhalf + j;
      if (slot0 < cnt) {
        int tok = lst[slot0];
        __builtin_nontemporal_store(acc0[i][j] + bias,
                                    &out[(size_t)tok * HDIM + f0 + col]);
      }
      if (slot1 < cnt) {
        int tok = lst[slot1];
        __builtin_nontemporal_store(acc1[i][j] + bias,
                                    &out[(size_t)tok * HDIM + f0 + col]);
      }
    }
  }
}

// ---------------------------------------------------------------- launch ----

extern "C" void kernel_launch(void* const* d_in, const int* in_sizes, int n_in,
                              void* d_out, int out_size, void* d_ws, size_t ws_size,
                              hipStream_t stream) {
  const float* tokens = (const float*)d_in[0];
  const float* gate_w = (const float*)d_in[1];
  const float* w1     = (const float*)d_in[2];
  const float* b1     = (const float*)d_in[3];
  const float* w2     = (const float*)d_in[4];
  const float* b2     = (const float*)d_in[5];
  float* out = (float*)d_out;

  const size_t W = (size_t)NEXP * HDIM * HDIM;   // 8M elems per weight tensor
  unsigned short* w1b = (unsigned short*)d_ws;   // 16 MB
  unsigned short* w2b = w1b + W;                 // 16 MB
  int* counts = (int*)(w2b + W);                 // 8 ints (pad to 16)
  int* lists  = counts + 16;                     // 8 * 16384 ints = 512 KB

  zero_counts_k<<<1, 32, 0, stream>>>(counts);
  gate_route_k<<<NTOK / 8, 256, 0, stream>>>(tokens, gate_w, counts, lists);
  cvt_weights_k<<<(int)(W / 4 / 256), 256, 0, stream>>>(w1, w2, w1b, w2b);
  moe_ffn_k<<<NEXP * TPE, 256, 0, stream>>>(tokens, w1b, b1, w2b, b2,
                                            lists, counts, out);
}
